// KernelSEBlock_10874857193773
// MI455X (gfx1250) — compile-verified, matrix operand order checked by reading, not run on hardware
//
#include <hip/hip_runtime.h>
#include <math.h>

// Problem constants (fixed by the reference):
//   B=8, C=128, H=W=64, K=3, KK=9, ck=1152, plane=4096
#define BD 8
#define CD 128
#define CK 1152
#define PLANE 4096
#define OUT_HALF 37748736  // 8*1152*4096

typedef __attribute__((ext_vector_type(16))) _Float16 v16h;
typedef __attribute__((ext_vector_type(8)))  float    v8f;

__device__ __forceinline__ float sigmoidf_(float x) {
    return 1.0f / (1.0f + expf(-x));
}

// ---------------------------------------------------------------------------
// Kernel A: per-(b,c) plane -> y[b, c*9+k] = mean over pixels of (conv+bias).
// Zero-padded conv mean = (sum_q wf[q] * T[q]) / 4096 + bf, where T[di][dj]
// is the plane sum restricted to pixels valid for that shift.
// One block per (b,c), 256 threads, float4 loads (4 pixels/thread/iter).
// ---------------------------------------------------------------------------
__global__ __launch_bounds__(256) void kA_mean(
    const float* __restrict__ x, const float* __restrict__ wf,
    const float* __restrict__ bf, float* __restrict__ y) {
    const int bc = blockIdx.x;            // 0..1023
    const int b = bc >> 7, c = bc & 127;
    const float4* plane4 = (const float4*)(x + (size_t)bc * PLANE);

    float acc[9];
    #pragma unroll
    for (int q = 0; q < 9; ++q) acc[q] = 0.f;

    #pragma unroll
    for (int it = 0; it < 4; ++it) {
        int i = threadIdx.x + it * 256;   // float4 index, 0..1023
        float4 v = plane4[i];
        int p = i * 4;
        int r = p >> 6, c0 = p & 63;      // 4 pixels share a row
        float sum4 = v.x + v.y + v.z + v.w;
        // column-restricted partial sums (4-wide groups only touch the
        // plane edge when c0==0 or c0==60):
        float sumA = sum4 - ((c0 == 60) ? v.w : 0.f);  // cols <= 62
        float sumC = sum4 - ((c0 == 0)  ? v.x : 0.f);  // cols >= 1
        bool r0 = (r <= 62), r2 = (r >= 1);
        acc[4] += sum4;
        acc[3] += sumA;
        acc[5] += sumC;
        acc[1] += r0 ? sum4 : 0.f;
        acc[0] += r0 ? sumA : 0.f;
        acc[2] += r0 ? sumC : 0.f;
        acc[7] += r2 ? sum4 : 0.f;
        acc[6] += r2 ? sumA : 0.f;
        acc[8] += r2 ? sumC : 0.f;
    }

    __shared__ float sm[256 * 9];
    #pragma unroll
    for (int q = 0; q < 9; ++q) sm[threadIdx.x * 9 + q] = acc[q];
    __syncthreads();

    __shared__ float T[9];
    if (threadIdx.x < 9) {
        float s = 0.f;
        for (int i = 0; i < 256; ++i) s += sm[i * 9 + threadIdx.x];
        T[threadIdx.x] = s;
    }
    __syncthreads();

    if (threadIdx.x < 9) {
        int k = threadIdx.x;
        int oc = c * 9 + k;
        const float* w = wf + oc * 9;
        float s = 0.f;
        #pragma unroll
        for (int q = 0; q < 9; ++q) s += w[q] * T[q];
        y[b * CK + oc] = s * (1.0f / 4096.0f) + bf[oc];
    }
}

// ---------------------------------------------------------------------------
// Kernel B: gate computation, single block of 288 threads (9 waves).
//  - fc_c branch (per channel 9->3->9) scalar.
//  - fc_k branch (per tap g: [8x128]@[128x8]) one wave per g using
//    v_wmma_f32_16x16x32_f16, K=128 as 4 chunks of 32, rows/cols 8..15 zero.
//    Operand packing is branchless: loads use clamped (always-valid)
//    addresses, zeros applied via selects, so no per-element exec juggling.
//  - gate[b][g*128+c] = ac[b,c,g] * ak[b,g,c]  (k-major flat order; the
//    reference's channel-shuffle mismatch is reproduced by indexing this
//    array with xf's channel index directly in kernel C).
// ---------------------------------------------------------------------------
__global__ __launch_bounds__(288) void kB_gate(
    const float* __restrict__ y, const float* __restrict__ wc1,
    const float* __restrict__ wc2, const float* __restrict__ wk1,
    const float* __restrict__ wk2, float* __restrict__ gate) {
    __shared__ float y_s[BD * CK];    // 9216
    __shared__ float ac_s[BD * CK];   // 9216, k-major per batch
    __shared__ float hk_s[9 * 64];    // [g][b][o]
    const int t = threadIdx.x;

    for (int i = t; i < BD * CK; i += 288) y_s[i] = y[i];
    __syncthreads();

    // ---- fc_c branch (scalar, tiny) ----
    for (int q = t; q < BD * CD; q += 288) {
        int b = q >> 7, c = q & 127;
        float yv[9];
        #pragma unroll
        for (int n = 0; n < 9; ++n) yv[n] = y_s[b * CK + c * 9 + n];
        float h[3];
        #pragma unroll
        for (int o = 0; o < 3; ++o) {
            float s = 0.f;
            #pragma unroll
            for (int n = 0; n < 9; ++n) s += wc1[(c * 3 + o) * 9 + n] * yv[n];
            h[o] = fmaxf(s, 0.f);
        }
        #pragma unroll
        for (int k = 0; k < 9; ++k) {
            float s = 0.f;
            #pragma unroll
            for (int o = 0; o < 3; ++o) s += h[o] * wc2[(c * 9 + k) * 3 + o];
            ac_s[b * CK + k * CD + c] = sigmoidf_(s);
        }
    }

    // ---- fc_k first matmul via WMMA: hk[g] = relu( Y_g(8x128) @ Wk1_g^T(128x8) ) ----
    {
        const int g = t >> 5;             // wave id = tap index 0..8
        const int lane = t & 31;
        const int half = lane >> 4;       // K-range selector per ISA layout
        const int idx = lane & 15;        // A: row M, B: col N
        const int m = idx & 7;            // clamped row/col (always valid)
        const bool valid = (idx < 8);
        const float* yrow = &y_s[m * CK];                 // A row m (b=m)
        const float4* brow4 = (const float4*)(wk1 + (g * 8 + m) * CD);
        v8f d = {};
        #pragma unroll
        for (int kc = 0; kc < 4; ++kc) {
            // B operand: K contiguous per row -> 4x float4 covers the
            // 16 halves ({half*8+0..7} and {16+half*8+0..7}).
            int kb0 = kc * 32 + half * 8;        // multiple of 8
            float4 q0 = brow4[kb0 / 4];
            float4 q1 = brow4[kb0 / 4 + 1];
            float4 q2 = brow4[kb0 / 4 + 4];      // +16 elements
            float4 q3 = brow4[kb0 / 4 + 5];
            float bl[16] = {q0.x, q0.y, q0.z, q0.w, q1.x, q1.y, q1.z, q1.w,
                            q2.x, q2.y, q2.z, q2.w, q3.x, q3.y, q3.z, q3.w};
            v16h a, bm;
            #pragma unroll
            for (int j = 0; j < 16; ++j) {
                // 16-bit A/B layout: halves 0..7 -> K = half*8 + j,
                //                    halves 8..15 -> K = 16 + half*8 + (j-8)
                int kl = (j < 8) ? (half * 8 + j) : (8 + half * 8 + j);
                int k = kc * 32 + kl;
                // A[m][k] = yk[b=m][g][c=k] = y[b, k*9+g]; rows 8..15 = 0
                float av = yrow[k * 9 + g];
                a[j]  = (_Float16)(valid ? av : 0.f);
                // B[k][n] = wk1[g, o=n, c=k]; cols 8..15 = 0
                bm[j] = (_Float16)(valid ? bl[j] : 0.f);
            }
            d = __builtin_amdgcn_wmma_f32_16x16x32_f16(
                false, a, false, bm, (short)0, d, false, false);
        }
        // D layout: reg r -> row m = r + 8*half, col n = idx.
        // Valid block is m<8 (half==0), n<8.
        if (half == 0 && idx < 8) {
            #pragma unroll
            for (int r = 0; r < 8; ++r)
                hk_s[g * 64 + r * 8 + idx] = fmaxf(d[r], 0.f);
        }
    }
    __syncthreads();

    // ---- ak + final gate ----
    for (int j = t; j < BD * CK; j += 288) {
        int b = j / CK, rem = j - b * CK;
        int g = rem >> 7, c = rem & 127;
        float s = 0.f;
        #pragma unroll
        for (int o = 0; o < 8; ++o)
            s += hk_s[g * 64 + b * 8 + o] * wk2[(g * CD + c) * 8 + o];
        gate[j] = ac_s[j] * sigmoidf_(s);
    }
}

// ---------------------------------------------------------------------------
// Kernel C: fused conv + gate multiply + both output halves. One block per
// (b,c): stage the 64x64 plane in LDS, recompute the 9-tap conv (cheaper
// than spilling/reloading 151 MB of xf). Each thread produces 4 consecutive
// pixels so every output store is a full-width global_store_b128 — this
// kernel is pure store-bandwidth (302 MB), so wide stores are what counts.
// ---------------------------------------------------------------------------
__global__ __launch_bounds__(256) void kC_out(
    const float* __restrict__ x, const float* __restrict__ wf,
    const float* __restrict__ bf, const float* __restrict__ gate,
    float* __restrict__ out) {
    const int bc = blockIdx.x;            // 0..1023
    const int b = bc >> 7, c = bc & 127;

    __shared__ __align__(16) float plane[PLANE];
    __shared__ float w_s[81];
    __shared__ float bf_s[9];
    __shared__ float g_s[9];

    {
        const float4* xp4 = (const float4*)(x + (size_t)bc * PLANE);
        float4* pl4 = (float4*)plane;
        for (int i = threadIdx.x; i < PLANE / 4; i += 256) pl4[i] = xp4[i];
    }
    if (threadIdx.x < 81) w_s[threadIdx.x] = wf[(size_t)(c * 9) * 9 + threadIdx.x];
    if (threadIdx.x < 9) {
        bf_s[threadIdx.x] = bf[c * 9 + threadIdx.x];
        // xf channel j = c*9+k gets gate_flat[b][j] (k-major array).
        g_s[threadIdx.x] = gate[b * CK + c * 9 + threadIdx.x];
    }
    __syncthreads();

    float* out1 = out + (size_t)(b * CK + c * 9) * PLANE;
    float* out2 = out1 + OUT_HALF;

    #pragma unroll
    for (int it = 0; it < 4; ++it) {
        int i = threadIdx.x + it * 256;   // group of 4 pixels, 0..1023
        int p = i * 4;
        int r = p >> 6, c0 = p & 63;      // 4 pixels share a row

        // 3x6 neighborhood covering pixels c0..c0+3 with halo.
        float nv[18];
        #pragma unroll
        for (int di = 0; di < 3; ++di) {
            int rr = r + di - 1;
            bool rok = (rr >= 0 && rr < 64);
            #pragma unroll
            for (int jj = 0; jj < 6; ++jj) {
                int cx = c0 - 1 + jj;
                bool ok = rok && (cx >= 0) && (cx < 64);
                nv[di * 6 + jj] = ok ? plane[rr * 64 + cx] : 0.f;
            }
        }

        #pragma unroll
        for (int k = 0; k < 9; ++k) {
            float s0 = bf_s[k], s1 = s0, s2 = s0, s3 = s0;
            #pragma unroll
            for (int di = 0; di < 3; ++di) {
                #pragma unroll
                for (int dj = 0; dj < 3; ++dj) {
                    float w = w_s[k * 9 + di * 3 + dj];
                    const float* row = &nv[di * 6 + dj];
                    s0 = fmaf(row[0], w, s0);
                    s1 = fmaf(row[1], w, s1);
                    s2 = fmaf(row[2], w, s2);
                    s3 = fmaf(row[3], w, s3);
                }
            }
            float gk = g_s[k];
            float4 o1 = {s0 * gk, s1 * gk, s2 * gk, s3 * gk};
            float4 o2 = {gk, gk, gk, gk};
            *(float4*)(out1 + k * PLANE + p) = o1;   // xf * gate
            *(float4*)(out2 + k * PLANE + p) = o2;   // attn broadcast
        }
    }
}

extern "C" void kernel_launch(void* const* d_in, const int* in_sizes, int n_in,
                              void* d_out, int out_size, void* d_ws, size_t ws_size,
                              hipStream_t stream) {
    const float* x   = (const float*)d_in[0];
    const float* wf  = (const float*)d_in[1];
    const float* bf  = (const float*)d_in[2];
    const float* wc1 = (const float*)d_in[3];
    const float* wc2 = (const float*)d_in[4];
    const float* wk1 = (const float*)d_in[5];
    const float* wk2 = (const float*)d_in[6];
    float* out = (float*)d_out;

    float* y    = (float*)d_ws;           // [8,1152]
    float* gate = y + BD * CK;            // [8,1152]; total ws use = 72 KB

    kA_mean<<<BD * CD, 256, 0, stream>>>(x, wf, bf, y);
    kB_gate<<<1, 288, 0, stream>>>(y, wc1, wc2, wk1, wk2, gate);
    kC_out <<<BD * CD, 256, 0, stream>>>(x, wf, bf, gate, out);
}
